// VQVAE_22247930593361
// MI455X (gfx1250) — matrix operand order repaired and checked by reading, not compiled, not run on hardware
//
#include <hip/hip_runtime.h>
#include <hip/hip_bf16.h>

// ---------------------------------------------------------------------------
// CDNA5 (gfx1250) VQ-VAE forward: every conv / deconv / codebook distance is
// an implicit GEMM on v_wmma_f32_16x16x32_bf16 (wave32).
//  - LDS fragments loaded as 16B vectors (ds_load_b128)
//  - weight tiles staged with GLOBAL_LOAD_ASYNC_TO_LDS_B128 when available
//  - double-buffered LDS tiles overlap staging with WMMA compute
// ---------------------------------------------------------------------------

typedef __attribute__((ext_vector_type(16))) __bf16 bf16x16;
typedef __attribute__((ext_vector_type(8)))  __bf16 bf16x8;
typedef __attribute__((ext_vector_type(8)))  float  f32x8;

#define ACT_NONE 0
#define ACT_RELU 1
#define ACT_TANH 2

#if __has_builtin(__builtin_amdgcn_global_load_async_to_lds_b128)
#define HAVE_ASYNC_LDS 1
#else
#define HAVE_ASYNC_LDS 0
#endif

#if HAVE_ASYNC_LDS
// builtin signature (probe-confirmed via diagnostic): (v4i AS1*, v4i AS3*, Imm, Imm)
typedef int v4i_ __attribute__((vector_size(16)));
typedef __attribute__((address_space(1))) v4i_* as1_v4i;
typedef __attribute__((address_space(3))) v4i_* as3_v4i;
#if __has_builtin(__builtin_amdgcn_s_wait_asynccnt)
#define WAIT_ASYNC() __builtin_amdgcn_s_wait_asynccnt(0)
#else
#define WAIT_ASYNC() asm volatile("s_wait_asynccnt 0x0" ::: "memory")
#endif
#endif

__device__ __forceinline__ __bf16 f2bf(float x) { return (__bf16)x; }

#define CAT16(a, b) __builtin_shufflevector(a, b, 0, 1, 2, 3, 4, 5, 6, 7, 8, 9, 10, 11, 12, 13, 14, 15)

// ---------------------------------------------------------------------------
// Generic implicit-GEMM convolution.
//   Y[b,co,ho,wo] = act( sum_{ci,kh,kw} Wt[co, ci*k*k+kh*k+kw] *
//                        X[b,ci,(ho*s-p+kh)/dil,(wo*s-p+kw)/dil] + bias[co] )
// lhs_dilation (dil>1) implements ConvTranspose as a conv over a dilated input.
// GEMM view: M = Cout, N = B*Hout*Wout, K = Cin*k*k.
// WG: 256 threads = 8 waves; tile 128(M) x 64(N); K-step 32; 2x LDS buffering.
// ---------------------------------------------------------------------------
__global__ __launch_bounds__(256) void conv_wmma(
    const float* __restrict__ X, const __bf16* __restrict__ Wt,
    const float* __restrict__ bias, float* __restrict__ Y,
    int B, int Cin, int Hin, int Win, int Cout, int Hout, int Wout,
    int ksz, int stride, int pad, int dil, int act)
{
  const int Ktot = Cin * ksz * ksz;
  const int HWo  = Hout * Wout;
  const int Ntot = B * HWo;
  const int n0   = blockIdx.x * 64;
  const int m0   = blockIdx.y * 128;

  __shared__ __align__(16) __bf16 As[2][128 * 32];   // A: [m][k] (weights)
  __shared__ __align__(16) __bf16 Bs[2][64 * 32];    // B: [n][k] (im2col patches)

  const int tid  = threadIdx.x;
  const int lane = tid & 31;
  const int wid  = tid >> 5;

  // async path legal only for full tiles with 64B-aligned rows
  const bool asyncA = ((Ktot & 31) == 0) && (m0 + 128 <= Cout);
  (void)asyncA;

  const int nK = (Ktot + 31) >> 5;
  f32x8 acc[4] = {};

  auto stageA = [&](int k0, int bsel) {
#if HAVE_ASYNC_LDS
    if (asyncA) {
      // 128x32 bf16 = 8 KB = 512 x b128; 2 async copies per thread
      for (int t = tid; t < 512; t += 256) {
        int m = t >> 2, ch = t & 3;
        const __bf16* src = Wt + (size_t)(m0 + m) * Ktot + k0 + ch * 8;
        __bf16* dst = &As[bsel][m * 32 + ch * 8];
        __builtin_amdgcn_global_load_async_to_lds_b128(
            (as1_v4i)(unsigned long long)(const void*)src,
            (as3_v4i)(unsigned int)(unsigned long long)(void*)dst,
            0, 0);
      }
      return;
    }
#endif
    for (int t = tid; t < 128 * 32; t += 256) {
      int m = t >> 5, kk = t & 31;
      int gm = m0 + m, gk = k0 + kk;
      __bf16 v = f2bf(0.0f);
      if (gm < Cout && gk < Ktot) v = Wt[gm * Ktot + gk];
      As[bsel][m * 32 + kk] = v;
    }
    if (k0 + 32 < Ktot) {
      int gm = m0 + (tid >> 1);
      if (gm < Cout) __builtin_prefetch(&Wt[gm * Ktot + k0 + 32], 0, 0);
    }
  };

  auto stageB = [&](int k0, int bsel) {
    for (int t = tid; t < 64 * 32; t += 256) {
      int nn = t >> 5, kk = t & 31;
      int gk = k0 + kk, gn = n0 + nn;
      float v = 0.0f;
      if (gk < Ktot && gn < Ntot) {
        int ci = gk / (ksz * ksz);
        int tt = gk - ci * ksz * ksz;
        int kh = tt / ksz, kw = tt - kh * ksz;
        int b  = gn / HWo;
        int rr = gn - b * HWo;
        int ho = rr / Wout, wo = rr - ho * Wout;
        int hl = ho * stride - pad + kh;
        int wl = wo * stride - pad + kw;
        if (dil == 1) {
          if (hl >= 0 && hl < Hin && wl >= 0 && wl < Win)
            v = X[((b * Cin + ci) * Hin + hl) * Win + wl];
        } else {
          if (hl >= 0 && wl >= 0 && (hl % dil) == 0 && (wl % dil) == 0) {
            int hi = hl / dil, wi = wl / dil;
            if (hi < Hin && wi < Win)
              v = X[((b * Cin + ci) * Hin + hi) * Win + wi];
          }
        }
      }
      Bs[bsel][nn * 32 + kk] = f2bf(v);
    }
  };

  stageA(0, 0);
  stageB(0, 0);
#if HAVE_ASYNC_LDS
  if (asyncA) WAIT_ASYNC();
#endif
  __syncthreads();

  for (int i = 0; i < nK; ++i) {
    const int cur = i & 1;
    if (i + 1 < nK) {            // overlap next-tile staging with compute
      stageA((i + 1) << 5, cur ^ 1);
      stageB((i + 1) << 5, cur ^ 1);
    }
    // ---- A fragment (16x32 bf16, ISA layout): two b128 LDS reads ----
    const int r   = wid * 16 + (lane & 15);
    const int kbA = (lane < 16) ? 0 : 8;
    bf16x8 a0 = *(const bf16x8*)&As[cur][r * 32 + kbA];
    bf16x8 a1 = *(const bf16x8*)&As[cur][r * 32 + 16 + kbA];
    bf16x16 af = CAT16(a0, a1);
    // ---- 4 WMMAs across the N tile ----
    const int kbB = (lane < 16) ? 0 : 8;
#pragma unroll
    for (int nt = 0; nt < 4; ++nt) {
      int c = nt * 16 + (lane & 15);
      bf16x8 b0 = *(const bf16x8*)&Bs[cur][c * 32 + kbB];
      bf16x8 b1 = *(const bf16x8*)&Bs[cur][c * 32 + 16 + kbB];
      bf16x16 bfm = CAT16(b0, b1);
      acc[nt] = __builtin_amdgcn_wmma_f32_16x16x32_bf16(
          false, af, false, bfm, (short)0, acc[nt], false, false);
    }
#if HAVE_ASYNC_LDS
    if (asyncA && i + 1 < nK) WAIT_ASYNC();
#endif
    __syncthreads();
  }

  // ---- epilogue: bias + activation, NCHW scatter ----
  const int mbase = m0 + wid * 16;
#pragma unroll
  for (int nt = 0; nt < 4; ++nt) {
    int gn = n0 + nt * 16 + (lane & 15);
    if (gn >= Ntot) continue;
    int b  = gn / HWo;
    int rr = gn - b * HWo;
#pragma unroll
    for (int v = 0; v < 8; ++v) {
      int co = mbase + ((lane < 16) ? v : v + 8);
      if (co >= Cout) continue;
      float r2 = acc[nt][v] + bias[co];
      if (act == ACT_RELU)      r2 = fmaxf(r2, 0.0f);
      else if (act == ACT_TANH) r2 = tanhf(r2);
      Y[(b * Cout + co) * HWo + rr] = r2;
    }
  }
}

// ---------------------------------------------------------------------------
// Weight prep: fp32 -> bf16, optional ConvTranspose transform
// (transpose in/out channels, flip kh/kw). Output flat (O, I*k*k) row-major.
// ---------------------------------------------------------------------------
__global__ void prep_w(const float* __restrict__ w, __bf16* __restrict__ o,
                       int O, int I, int k, int tio, int flip, int total)
{
  int idx = blockIdx.x * blockDim.x + threadIdx.x;
  if (idx >= total) return;
  int kk  = k * k;
  int oc  = idx / (I * kk);
  int rem = idx - oc * I * kk;
  int ic  = rem / kk;
  int t   = rem - ic * kk;
  int kh  = t / k, kw = t - kh * k;
  if (flip) { kh = k - 1 - kh; kw = k - 1 - kw; }
  int src = tio ? ((ic * O + oc) * kk + kh * k + kw)
                : ((oc * I + ic) * kk + kh * k + kw);
  o[idx] = f2bf(w[src]);
}

// Codebook prep: bf16 copy + per-code squared norm. One block per code.
__global__ void cb_prep(const float* __restrict__ cb, __bf16* __restrict__ cbb,
                        float* __restrict__ cbn)
{
  int kc = blockIdx.x;
  int d  = threadIdx.x;  // 128 threads
  float v = cb[kc * 128 + d];
  cbb[kc * 128 + d] = f2bf(v);
  __shared__ float s[128];
  s[d] = v * v;
  __syncthreads();
  for (int st = 64; st > 0; st >>= 1) {
    if (d < st) s[d] += s[d + st];
    __syncthreads();
  }
  if (d == 0) cbn[kc] = s[0];
}

// ---------------------------------------------------------------------------
// BatchNorm: per-channel mean/var over (N,H,W), then in-place normalize.
// ---------------------------------------------------------------------------
__global__ __launch_bounds__(256) void bn_stats(const float* __restrict__ X,
                                                float* __restrict__ mean,
                                                float* __restrict__ var,
                                                int Bn, int C, int HW)
{
  int c   = blockIdx.x;
  int tid = threadIdx.x;
  __shared__ float s1[256], s2[256];
  float a = 0.0f, b = 0.0f;
  int tot = Bn * HW;
  for (int j = tid; j < tot; j += 256) {
    int n = j / HW, i = j - n * HW;
    float v = X[(n * C + c) * HW + i];
    a += v; b += v * v;
  }
  s1[tid] = a; s2[tid] = b;
  __syncthreads();
  for (int st = 128; st > 0; st >>= 1) {
    if (tid < st) { s1[tid] += s1[tid + st]; s2[tid] += s2[tid + st]; }
    __syncthreads();
  }
  if (tid == 0) {
    float m = s1[0] / (float)tot;
    mean[c] = m;
    var[c]  = s2[0] / (float)tot - m * m;
  }
}

__global__ void bn_apply(float* __restrict__ x, const float* __restrict__ mean,
                         const float* __restrict__ var,
                         const float* __restrict__ g, const float* __restrict__ b,
                         int C, int HW, int total)
{
  int i = blockIdx.x * blockDim.x + threadIdx.x;
  if (i >= total) return;
  int c = (i / HW) % C;
  float inv = rsqrtf(var[c] + 1e-5f);
  x[i] = (x[i] - mean[c]) * inv * g[c] + b[c];
}

// ---------------------------------------------------------------------------
// GroupNorm (8 groups of 16 channels) stats per (n, group), then
// normalize * gamma + beta with fused tanh -> z_e (written straight to d_out).
// ---------------------------------------------------------------------------
__global__ __launch_bounds__(256) void gn_stats(const float* __restrict__ X,
                                                float* __restrict__ mean,
                                                float* __restrict__ var)
{
  int ng  = blockIdx.x;            // 32*8 = 256
  int n   = ng >> 3, g = ng & 7;
  int tid = threadIdx.x;
  __shared__ float s1[256], s2[256];
  float a = 0.0f, b = 0.0f;
  const int HW = 1024, Cg = 16, C = 128;
  for (int j = tid; j < Cg * HW; j += 256) {
    int c = g * Cg + j / HW;
    int i = j & (HW - 1);
    float v = X[(n * C + c) * HW + i];
    a += v; b += v * v;
  }
  s1[tid] = a; s2[tid] = b;
  __syncthreads();
  for (int st = 128; st > 0; st >>= 1) {
    if (tid < st) { s1[tid] += s1[tid + st]; s2[tid] += s2[tid + st]; }
    __syncthreads();
  }
  if (tid == 0) {
    float m = s1[0] / 16384.0f;
    mean[ng] = m;
    var[ng]  = s2[0] / 16384.0f - m * m;
  }
}

__global__ void gn_apply_tanh(const float* __restrict__ x, float* __restrict__ out,
                              const float* __restrict__ mean, const float* __restrict__ var,
                              const float* __restrict__ g, const float* __restrict__ b,
                              int total)
{
  int i = blockIdx.x * blockDim.x + threadIdx.x;
  if (i >= total) return;
  int n  = i / 131072;           // 128*1024 elems per image
  int c  = (i / 1024) & 127;
  int ng = n * 8 + (c >> 4);
  float inv = rsqrtf(var[ng] + 1e-5f);
  out[i] = tanhf((x[i] - mean[ng]) * inv * g[c] + b[c]);
}

__global__ void add_inplace(float* __restrict__ y, const float* __restrict__ s, int n)
{
  int i = blockIdx.x * blockDim.x + threadIdx.x;
  if (i < n) y[i] += s[i];
}

// ---------------------------------------------------------------------------
// Vector quantization: 32768 rows (D=128) against 1024 codes.
// dist = ||cb||^2 - 2 e.cb  (||e||^2 constant per row -> dropped for argmin).
// Block = 128 threads (4 waves); wave owns 16 rows; A-frags resident in VGPRs
// across the whole code loop; codebook tiles staged in LDS (b128 frag reads).
// ---------------------------------------------------------------------------
__global__ __launch_bounds__(128) void vq_wmma(
    const float* __restrict__ ze, const __bf16* __restrict__ cbb,
    const float* __restrict__ cbn, const float* __restrict__ cbf,
    float* __restrict__ zq_out, float* __restrict__ zq_buf)
{
  const int tid  = threadIdx.x;
  const int lane = tid & 31;
  const int wid  = tid >> 5;
  const int rbase = blockIdx.x * 64 + wid * 16;

  __shared__ __align__(16) __bf16 cbt[16 * 128];   // 16 codes x 128 dims
  __shared__ float  redv[4][32][8];
  __shared__ int    redi[4][32][8];
  __shared__ int    rowidx[64];

  // A fragments: 16 rows x 128 dims = 4 K-steps, resident in VGPRs
  bf16x16 af[4];
  {
    int r  = rbase + (lane & 15);
    int b  = r >> 10;
    int hw = r & 1023;
    const float* zr = ze + (long)b * 131072 + hw;   // dim d at zr[d*1024]
    int kb = (lane < 16) ? 0 : 8;
    for (int ks = 0; ks < 4; ++ks) {
#pragma unroll
      for (int v = 0; v < 8; ++v) {
        int kq = ks * 32 + ((v < 4) ? (kb + 2 * v) : (16 + kb + 2 * (v - 4)));
        af[ks][2 * v]     = f2bf(zr[kq * 1024]);
        af[ks][2 * v + 1] = f2bf(zr[(kq + 1) * 1024]);
      }
    }
  }

  float minv[8]; int mini[8];
#pragma unroll
  for (int v = 0; v < 8; ++v) { minv[v] = 3.4e38f; mini[v] = 0; }

  const int col = lane & 15;
  const int kb  = (lane < 16) ? 0 : 8;
  for (int ct = 0; ct < 64; ++ct) {     // 64 tiles of 16 codes
    __syncthreads();
    for (int t = tid; t < 16 * 128; t += 128) cbt[t] = cbb[ct * 2048 + t];
    __syncthreads();

    f32x8 acc = {};
#pragma unroll
    for (int ks = 0; ks < 4; ++ks) {
      const __bf16* base = &cbt[col * 128 + ks * 32 + kb];
      bf16x8 c0 = *(const bf16x8*)base;
      bf16x8 c1 = *(const bf16x8*)(base + 16);
      bf16x16 bfm = CAT16(c0, c1);
      acc = __builtin_amdgcn_wmma_f32_16x16x32_bf16(
          false, af[ks], false, bfm, (short)0, acc, false, false);
    }
    int code = ct * 16 + col;
    float cn = cbn[code];
#pragma unroll
    for (int v = 0; v < 8; ++v) {
      float d = cn - 2.0f * acc[v];
      if (d < minv[v]) { minv[v] = d; mini[v] = code; }
    }
  }

#pragma unroll
  for (int v = 0; v < 8; ++v) { redv[wid][lane][v] = minv[v]; redi[wid][lane][v] = mini[v]; }
  __syncthreads();

  // one thread per row: reduce over the 16 lanes that share it
  if (tid < 64) {
    int w  = tid >> 4;
    int m  = tid & 15;
    int v  = m & 7;
    int lo = (m < 8) ? 0 : 16;
    float bv = 3.4e38f; int bi = 0;
    for (int l = 0; l < 16; ++l) {
      float d = redv[w][lo + l][v];
      if (d < bv) { bv = d; bi = redi[w][lo + l][v]; }
    }
    rowidx[tid] = bi;
  }
  __syncthreads();

  // gather z_q rows (fp32 codebook) into NCHW layout, twice
  for (int t = tid; t < 64 * 128; t += 128) {
    int rloc = t >> 7, d = t & 127;
    int r  = blockIdx.x * 64 + rloc;
    int b  = r >> 10;
    int hw = r & 1023;
    float v = cbf[rowidx[rloc] * 128 + d];
    long o = (long)b * 131072 + (long)d * 1024 + hw;
    zq_out[o] = v;
    zq_buf[o] = v;
  }
}

// ---------------------------------------------------------------------------
// Host-side orchestration
// ---------------------------------------------------------------------------
extern "C" void kernel_launch(void* const* d_in, const int* in_sizes, int n_in,
                              void* d_out, int out_size, void* d_ws, size_t ws_size,
                              hipStream_t stream)
{
  (void)in_sizes; (void)n_in; (void)out_size; (void)ws_size;

  // ---- inputs (setup_inputs dict order) ----
  const float* x        = (const float*)d_in[0];
  const float* e_w1     = (const float*)d_in[1];
  const float* e_b1     = (const float*)d_in[2];
  const float* e_bn1_g  = (const float*)d_in[3];
  const float* e_bn1_b  = (const float*)d_in[4];
  const float* e_w2     = (const float*)d_in[5];
  const float* e_b2     = (const float*)d_in[6];
  const float* e_bn2_g  = (const float*)d_in[7];
  const float* e_bn2_b  = (const float*)d_in[8];
  const float* rW1[4], *rB1[4], *rW2[4], *rB2[4];   // e_r1, e_r2, d_r1, d_r2
  for (int i = 0; i < 4; ++i) {
    rW1[i] = (const float*)d_in[9 + 4 * i + 0];
    rB1[i] = (const float*)d_in[9 + 4 * i + 1];
    rW2[i] = (const float*)d_in[9 + 4 * i + 2];
    rB2[i] = (const float*)d_in[9 + 4 * i + 3];
  }
  const float* gn_g     = (const float*)d_in[25];
  const float* gn_b     = (const float*)d_in[26];
  const float* codebook = (const float*)d_in[27];
  const float* d_w1     = (const float*)d_in[28];
  const float* d_b1     = (const float*)d_in[29];
  const float* dt1_w    = (const float*)d_in[30];
  const float* dt1_b    = (const float*)d_in[31];
  const float* d_bn1_g  = (const float*)d_in[32];
  const float* d_bn1_b  = (const float*)d_in[33];
  const float* dt2_w    = (const float*)d_in[34];
  const float* dt2_b    = (const float*)d_in[35];
  const float* d_bn2_g  = (const float*)d_in[36];
  const float* d_bn2_b  = (const float*)d_in[37];
  const float* dt3_w    = (const float*)d_in[38];
  const float* dt3_b    = (const float*)d_in[39];

  // ---- outputs: (x_hat, z_e, z_q) concatenated ----
  float* out  = (float*)d_out;
  float* xhat = out;                 // 32*3*128*128  = 1,572,864
  float* ze   = out + 1572864;       // 32*128*32*32  = 4,194,304
  float* zq   = out + 5767168;       // 32*128*32*32  = 4,194,304

  // ---- workspace carve-up ----
  float* A     = (float*)d_ws;       // 16,777,216 f32 (64x64 acts; aliased below)
  float* Bb    = A + 16777216;       //  4,194,304 f32 (32x32 acts)
  float* stats = Bb + 4194304;       //  512 f32 (mean|var for BN/GN)
  float* cbn   = stats + 512;        //  1024 f32 (codebook norms)
  __bf16* wp   = (__bf16*)(cbn + 1024);     // 16B-aligned
  __bf16* wE1 = wp; wp += 3456;      // 128 x 27
  __bf16* wE2 = wp; wp += 147456;    // 128 x 1152
  __bf16* wR1[4], *wR2[4];
  for (int i = 0; i < 4; ++i) { wR1[i] = wp; wp += 147456; wR2[i] = wp; wp += 147456; }
  __bf16* wD1 = wp; wp += 16384;     // 128 x 128
  __bf16* wT1 = wp; wp += 16384;
  __bf16* wT2 = wp; wp += 147456;
  __bf16* wT3 = wp; wp += 3456;      // 3 x 1152
  __bf16* cbb = wp; wp += 131072;    // 1024 x 128
  // 32x32 temporaries aliased into A (dead while resblocks / decoder head run)
  float* Cc = A;
  float* Ee = A + 4194304;
  float* Ff = A + 8388608;

  auto ew = [&](int total) { return dim3((total + 255) / 256); };

  // ---- weight / codebook prep ----
  prep_w<<<ew(3456),   256, 0, stream>>>(e_w1, wE1, 128, 3,   3, 0, 0, 3456);
  prep_w<<<ew(147456), 256, 0, stream>>>(e_w2, wE2, 128, 128, 3, 0, 0, 147456);
  for (int i = 0; i < 4; ++i) {
    prep_w<<<ew(147456), 256, 0, stream>>>(rW1[i], wR1[i], 128, 128, 3, 0, 0, 147456);
    prep_w<<<ew(147456), 256, 0, stream>>>(rW2[i], wR2[i], 128, 128, 3, 0, 0, 147456);
  }
  prep_w<<<ew(16384),  256, 0, stream>>>(d_w1,  wD1, 128, 128, 1, 0, 0, 16384);
  prep_w<<<ew(16384),  256, 0, stream>>>(dt1_w, wT1, 128, 128, 1, 1, 0, 16384);
  prep_w<<<ew(147456), 256, 0, stream>>>(dt2_w, wT2, 128, 128, 3, 1, 1, 147456);
  prep_w<<<ew(3456),   256, 0, stream>>>(dt3_w, wT3, 3,   128, 3, 1, 1, 3456);
  cb_prep<<<1024, 128, 0, stream>>>(codebook, cbb, cbn);

  auto convL = [&](const float* Xp, const __bf16* W, const float* bias, float* Y,
                   int Bn, int Cin, int Hin, int Win, int Cout, int Hout, int Wout,
                   int k, int s, int p, int dil, int act) {
    int Ntot = Bn * Hout * Wout;
    dim3 g((Ntot + 63) / 64, (Cout + 127) / 128);
    conv_wmma<<<g, 256, 0, stream>>>(Xp, W, bias, Y, Bn, Cin, Hin, Win,
                                     Cout, Hout, Wout, k, s, p, dil, act);
  };

  // ================= encoder =================
  convL(x, wE1, e_b1, A, 32, 3, 128, 128, 128, 64, 64, 3, 2, 1, 1, ACT_RELU);
  bn_stats<<<128, 256, 0, stream>>>(A, stats, stats + 128, 32, 128, 4096);
  bn_apply<<<ew(16777216), 256, 0, stream>>>(A, stats, stats + 128, e_bn1_g, e_bn1_b,
                                             128, 4096, 16777216);
  convL(A, wE2, e_b2, Bb, 32, 128, 64, 64, 128, 32, 32, 3, 2, 1, 1, ACT_RELU);
  bn_stats<<<128, 256, 0, stream>>>(Bb, stats, stats + 128, 32, 128, 1024);
  bn_apply<<<ew(4194304), 256, 0, stream>>>(Bb, stats, stats + 128, e_bn2_g, e_bn2_b,
                                            128, 1024, 4194304);
  for (int i = 0; i < 2; ++i) {      // e_r1, e_r2
    convL(Bb, wR1[i], rB1[i], Cc, 32, 128, 32, 32, 128, 32, 32, 3, 1, 1, 1, ACT_RELU);
    convL(Cc, wR2[i], rB2[i], Ee, 32, 128, 32, 32, 128, 32, 32, 3, 1, 1, 1, ACT_NONE);
    add_inplace<<<ew(4194304), 256, 0, stream>>>(Bb, Ee, 4194304);
  }
  gn_stats<<<256, 256, 0, stream>>>(Bb, stats, stats + 256);
  gn_apply_tanh<<<ew(4194304), 256, 0, stream>>>(Bb, ze, stats, stats + 256,
                                                 gn_g, gn_b, 4194304);

  // ================= quantize =================
  vq_wmma<<<512, 128, 0, stream>>>(ze, cbb, cbn, codebook, zq, Bb);

  // ================= decoder =================
  convL(Bb, wD1, d_b1, Cc, 32, 128, 32, 32, 128, 32, 32, 1, 1, 0, 1, ACT_NONE);
  for (int i = 2; i < 4; ++i) {      // d_r1, d_r2
    convL(Cc, wR1[i], rB1[i], Ee, 32, 128, 32, 32, 128, 32, 32, 3, 1, 1, 1, ACT_RELU);
    convL(Ee, wR2[i], rB2[i], Ff, 32, 128, 32, 32, 128, 32, 32, 3, 1, 1, 1, ACT_NONE);
    add_inplace<<<ew(4194304), 256, 0, stream>>>(Cc, Ff, 4194304);
  }
  convL(Cc, wT1, dt1_b, Bb, 32, 128, 32, 32, 128, 32, 32, 1, 1, 0, 1, ACT_RELU);
  bn_stats<<<128, 256, 0, stream>>>(Bb, stats, stats + 128, 32, 128, 1024);
  bn_apply<<<ew(4194304), 256, 0, stream>>>(Bb, stats, stats + 128, d_bn1_g, d_bn1_b,
                                            128, 1024, 4194304);
  convL(Bb, wT2, dt2_b, A, 32, 128, 32, 32, 128, 64, 64, 3, 1, 1, 2, ACT_RELU);
  bn_stats<<<128, 256, 0, stream>>>(A, stats, stats + 128, 32, 128, 4096);
  bn_apply<<<ew(16777216), 256, 0, stream>>>(A, stats, stats + 128, d_bn2_g, d_bn2_b,
                                             128, 4096, 16777216);
  convL(A, wT3, dt3_b, xhat, 32, 128, 64, 64, 3, 128, 128, 3, 1, 1, 2, ACT_TANH);
}
